// attn_mh_86131274154337
// MI455X (gfx1250) — matrix operand order repaired and checked by reading, not compiled, not run on hardware
//
#include <hip/hip_runtime.h>
#include <cstdint>
#include <cstddef>

// ---------------------------------------------------------------------------
// MI455X (gfx1250) fused attention block, bf16 WMMA + Tensor Data Mover path.
//   B=1, S=4096, E=1024, H=8, DK=128
// ---------------------------------------------------------------------------

typedef __bf16 bf16;
typedef unsigned int u32;
typedef __attribute__((ext_vector_type(16))) __bf16 v16bf;
typedef __attribute__((ext_vector_type(8)))  float  v8f;
typedef __attribute__((ext_vector_type(4)))  unsigned int v4u;
typedef __attribute__((ext_vector_type(8)))  int v8i;
typedef __attribute__((ext_vector_type(4)))  int v4i;

#define S_LEN  4096
#define EDIM   1024
#define NHEAD  8
#define DKDIM  128
#define LOG2E  1.4426950408889634f

#if defined(__has_builtin)
#if __has_builtin(__builtin_amdgcn_tensor_load_to_lds)
#define HAVE_TDM 1
#endif
#endif

static __device__ __forceinline__ v8f zero8() {
  v8f z = {0.f, 0.f, 0.f, 0.f, 0.f, 0.f, 0.f, 0.f};
  return z;
}

static __device__ __forceinline__ v8f wmma_bf16(v16bf a, v16bf b, v8f c) {
  // D = A(16x32 bf16) * B(32x16 bf16) + C(16x16 f32)
  return __builtin_amdgcn_wmma_f32_16x16x32_bf16(
      /*neg_a=*/false, a, /*neg_b=*/false, b,
      /*c_mod=*/(short)0, c, /*reuse_a=*/false, /*reuse_b=*/false);
}

// A-fragment K index for VGPR pair i (elements 2i,2i+1), contiguous order:
// lanes 0-15 cover K {0..15}, lanes 16-31 cover K {16..31}.
static __device__ __forceinline__ int a_kbase(int i, int half) {
  return (i << 1) + (half << 4);
}

#ifdef HAVE_TDM
// TDM: DMA a [rows x 128] bf16 tile (row stride 128 elem) from global -> LDS.
// D# packing per CDNA5 ISA ch.8: group0 {count, lds_addr, global_addr, type=2},
// group1 {data_size=2B, tensor_dim0/1, tile_dim0/1, dim0_stride}. 2-D tile so
// groups 2/3 are zero.  Tracked on TENSORcnt (in-order per wave).
static __device__ __forceinline__ void tdm_load_tile(u32 lds_off,
                                                     const bf16* gptr,
                                                     u32 rows) {
  unsigned long long ga = (unsigned long long)(const void*)gptr;
  v4u g0;
  g0[0] = 1u;                                           // count=1, user mode
  g0[1] = lds_off;                                      // lds_addr (bytes)
  g0[2] = (u32)(ga & 0xffffffffu);                      // global_addr lo
  g0[3] = (u32)((ga >> 32) & 0x1ffffffu) | (2u << 30);  // addr hi | type=2
  v8i g1;
  g1[0] = (int)(1u << 16);            // workgroup_mask=0 | data_size=1 (2B)
  g1[1] = (int)((u32)DKDIM << 16);    // atomic_barrier=0 | tensor_dim0 lo16
  g1[2] = (int)((u32)S_LEN << 16);    // tensor_dim0 hi=0 | tensor_dim1 lo16
  g1[3] = (int)((u32)DKDIM << 16);    // tensor_dim1 hi=0 | tile_dim0=128
  g1[4] = (int)rows;                  // tile_dim1 | tile_dim2=0
  g1[5] = (int)DKDIM;                 // tensor_dim0_stride lo32
  g1[6] = 0;                          // stride hi | tensor_dim1_stride lo
  g1[7] = 0;
  v4i gz = {0, 0, 0, 0};
#if defined(__clang_major__) && __clang_major__ >= 23
  v8i gz8 = {0, 0, 0, 0, 0, 0, 0, 0};
  __builtin_amdgcn_tensor_load_to_lds(g0, g1, gz, gz, gz8, 0);
#else
  __builtin_amdgcn_tensor_load_to_lds(g0, g1, gz, gz, 0);
#endif
}
#endif

// ---------------------------------------------------------------------------
// Kernel 1: input projection  C[s,o] = X[s,:] . W[o,:] + b[o]
// Output bf16 head-major [H][S][DK].
// ---------------------------------------------------------------------------
__global__ __launch_bounds__(128) void proj_kernel(
    const float* __restrict__ X, const float* __restrict__ W,
    const float* __restrict__ bias, bf16* __restrict__ out) {
  const int lane = threadIdx.x & 31;
  const int wave = threadIdx.x >> 5;
  const int n16  = lane & 15;
  const int half = lane >> 4;
  const int s0   = blockIdx.y * 64 + wave * 16;
  const int o0   = blockIdx.x * 64;

  v8f acc[4];
#pragma unroll
  for (int nt = 0; nt < 4; ++nt) acc[nt] = zero8();

  for (int k0 = 0; k0 < EDIM; k0 += 32) {
    v16bf a;
    const float* xp = X + (size_t)(s0 + n16) * EDIM + k0;
#pragma unroll
    for (int i = 0; i < 8; ++i) {
      int k = a_kbase(i, half);
      a[2 * i]     = (bf16)xp[k];
      a[2 * i + 1] = (bf16)xp[k + 1];
    }
#pragma unroll
    for (int nt = 0; nt < 4; ++nt) {
      v16bf b;
      const float* wp = W + (size_t)(o0 + nt * 16 + n16) * EDIM + k0 + half * 16;
#pragma unroll
      for (int i = 0; i < 8; ++i) {
        b[2 * i]     = (bf16)wp[2 * i];
        b[2 * i + 1] = (bf16)wp[2 * i + 1];
      }
      acc[nt] = wmma_bf16(a, b, acc[nt]);
    }
  }

#pragma unroll
  for (int nt = 0; nt < 4; ++nt) {
    int o = o0 + nt * 16 + n16;
    int h = o >> 7;
    int d = o & 127;
    float bv = bias[o];
#pragma unroll
    for (int r = 0; r < 8; ++r) {
      int s = s0 + r + 8 * half;
      out[((size_t)h * S_LEN + s) * DKDIM + d] = (bf16)(acc[nt][r] + bv);
    }
  }
}

// ---------------------------------------------------------------------------
// Kernel 2: flash attention, mask + post-softmax group_prob scaling.
// Block = 128 thr (4 waves) = 1 head x 64 queries.
//  - K tiles (32x128) DMA'd by the Tensor Data Mover, double-buffered;
//    wave 0 issues next chunk then s_wait_tensorcnt(1) for the current one.
//  - V tile transposed into LDS with a row-per-lane mapping (conflict-free
//    2B writes), so PV B-fragments are contiguous 32B ds reads.
// ---------------------------------------------------------------------------
__global__ __launch_bounds__(128) void attn_kernel(
    const bf16* __restrict__ Qh, const bf16* __restrict__ Kh,
    const bf16* __restrict__ Vh, const int* __restrict__ mask,
    const float* __restrict__ gp, bf16* __restrict__ out /* [S][E] */) {
  __shared__ __align__(32) bf16 ldsK[2][32 * DKDIM];  // [key][d] row-major
  __shared__ __align__(32) bf16 ldsVt[DKDIM * 32];    // [d][key] transposed
  __shared__ __align__(32) bf16 ldsP[4][16 * 32];     // per-wave P' staging

  const int head = blockIdx.x;
  const int lane = threadIdx.x & 31;
  const int wave = threadIdx.x >> 5;
  const int n16  = lane & 15;
  const int half = lane >> 4;
  const int q0   = blockIdx.y * 64 + wave * 16;

  const bf16* Qbase = Qh + (size_t)head * S_LEN * DKDIM;
  const bf16* Kbase = Kh + (size_t)head * S_LEN * DKDIM;
  const bf16* Vbase = Vh + (size_t)head * S_LEN * DKDIM;

  // Preload Q A-fragments for the 4 K=32 slices of DK=128 (VGPR-resident)
  v16bf qf[4];
#pragma unroll
  for (int c = 0; c < 4; ++c) {
    const bf16* qp = Qbase + (size_t)(q0 + n16) * DKDIM + c * 32;
#pragma unroll
    for (int i = 0; i < 8; ++i) {
      int k = a_kbase(i, half);
      qf[c][2 * i]     = qp[k];
      qf[c][2 * i + 1] = qp[k + 1];
    }
  }

  v8f acc[8];
#pragma unroll
  for (int dt = 0; dt < 8; ++dt) acc[dt] = zero8();
  float mrow[8], lrow[8];
#pragma unroll
  for (int r = 0; r < 8; ++r) { mrow[r] = -3.0e38f; lrow[r] = 0.f; }

  const float sscale = 0.08838834764831845f * LOG2E;  // 1/sqrt(128)*log2(e)

#ifdef HAVE_TDM
  const u32 ldsK_off0 = (u32)(unsigned long long)(uintptr_t)&ldsK[0][0];
  const u32 ldsK_off1 = (u32)(unsigned long long)(uintptr_t)&ldsK[1][0];
  if (wave == 0) tdm_load_tile(ldsK_off0, Kbase, 32);  // prologue chunk 0
#endif

  for (int t0 = 0; t0 < S_LEN; t0 += 32) {
    const int cur = (t0 >> 5) & 1;
    __syncthreads();  // (A) all waves done reading ldsVt and ldsK[cur^1]

    // ---- stage V transposed: row-per-lane mapping, conflict-free writes ----
    {
      const int tid = threadIdx.x;
#pragma unroll
      for (int it = 0; it < 4; ++it) {
        int idx  = tid + it * 128;     // 0..511
        int row  = idx & 31;           // key (per-lane -> 2B write stride)
        int col8 = (idx >> 5) * 8;     // d offset 0..120
        const uint4 raw = *(const uint4*)(Vbase + (size_t)(t0 + row) * DKDIM + col8);
        const bf16* rp = (const bf16*)&raw;
#pragma unroll
        for (int e = 0; e < 8; ++e)
          ldsVt[(col8 + e) * 32 + row] = rp[e];
      }
    }

#ifdef HAVE_TDM
    if (wave == 0) {
      if (t0 + 32 < S_LEN) {
        tdm_load_tile(cur ? ldsK_off0 : ldsK_off1,
                      Kbase + (size_t)(t0 + 32) * DKDIM, 32);
        __builtin_amdgcn_s_wait_tensorcnt((short)1);  // current tile landed
      } else {
        __builtin_amdgcn_s_wait_tensorcnt((short)0);
      }
    }
#else
    {  // fallback: manual K staging
      const int tid = threadIdx.x;
#pragma unroll
      for (int it = 0; it < 4; ++it) {
        int idx  = tid + it * 128;
        int row  = idx >> 4;
        int col8 = (idx & 15) * 8;
        *(uint4*)(&ldsK[cur][row * DKDIM + col8]) =
            *(const uint4*)(Kbase + (size_t)(t0 + row) * DKDIM + col8);
      }
    }
#endif
    __syncthreads();  // (B) V stores + K tensor-DMA visible to all waves

    // ---- scores S = Q Kt for two 16-key subtiles (8 WMMA) ----
    v8f sc[2];
    sc[0] = zero8(); sc[1] = zero8();
#pragma unroll
    for (int sub = 0; sub < 2; ++sub) {
#pragma unroll
      for (int c = 0; c < 4; ++c) {
        const v16bf b = *(const v16bf*)
            &ldsK[cur][(sub * 16 + n16) * DKDIM + c * 32 + half * 16];
        sc[sub] = wmma_bf16(qf[c], b, sc[sub]);
      }
    }

    // ---- mask + scale into log2 domain ----
    float s2[2][8];
#pragma unroll
    for (int sub = 0; sub < 2; ++sub) {
      int t = t0 + sub * 16 + n16;
#pragma unroll
      for (int r = 0; r < 8; ++r) {
        int q = q0 + r + 8 * half;
        int mk = mask[(size_t)q * S_LEN + t];
        float v = sc[sub][r] * sscale;
        s2[sub][r] = (mk != 0) ? v : (-1e9f * LOG2E);  // mirrors reference
      }
    }

    // ---- online softmax + P' = p * group_prob staged to LDS ----
#pragma unroll
    for (int r = 0; r < 8; ++r) {
      float cm = fmaxf(s2[0][r], s2[1][r]);
#pragma unroll
      for (int off = 1; off < 16; off <<= 1)
        cm = fmaxf(cm, __shfl_xor(cm, off, 32));  // within 16-lane half
      float mn   = fmaxf(mrow[r], cm);
      float corr = exp2f(mrow[r] - mn);
      mrow[r] = mn;
      float p0 = exp2f(s2[0][r] - mn);
      float p1 = exp2f(s2[1][r] - mn);
      float ps = p0 + p1;
#pragma unroll
      for (int off = 1; off < 16; off <<= 1)
        ps += __shfl_xor(ps, off, 32);
      lrow[r] = lrow[r] * corr + ps;
#pragma unroll
      for (int dt = 0; dt < 8; ++dt) acc[dt][r] *= corr;

      int q = q0 + r + 8 * half;
      float g0 = gp[(size_t)q * S_LEN + t0 + n16];
      float g1 = gp[(size_t)q * S_LEN + t0 + 16 + n16];
      int prow = (r + 8 * half) * 32;
      ldsP[wave][prow + n16]      = (bf16)(p0 * g0);
      ldsP[wave][prow + 16 + n16] = (bf16)(p1 * g1);
    }

    // ---- A fragment for PV from staged P' ----
    v16bf pa;
    {
      const bf16* pp = &ldsP[wave][n16 * 32];
#pragma unroll
      for (int i = 0; i < 8; ++i) {
        int k = a_kbase(i, half);
        pa[2 * i]     = pp[k];
        pa[2 * i + 1] = pp[k + 1];
      }
    }
    // ---- acc += P'(16x32) V(32x128)  (8 WMMA) ----
#pragma unroll
    for (int dt = 0; dt < 8; ++dt) {
      const v16bf b = *(const v16bf*)&ldsVt[(dt * 16 + n16) * 32 + half * 16];
      acc[dt] = wmma_bf16(pa, b, acc[dt]);
    }
  }

  // ---- finalize z = acc / l ; store bf16 in [s][h*128 + d] ----
#pragma unroll
  for (int r = 0; r < 8; ++r) {
    float inv = 1.f / fmaxf(lrow[r], 1e-30f);
    int s = q0 + r + 8 * half;
#pragma unroll
    for (int dt = 0; dt < 8; ++dt) {
      out[(size_t)s * EDIM + head * DKDIM + dt * 16 + n16] =
          (bf16)(acc[dt][r] * inv);
    }
  }
}

// ---------------------------------------------------------------------------
// Kernel 3: output projection + bias + residual: Y = Z.Wo^T + bo + X  (f32)
// ---------------------------------------------------------------------------
__global__ __launch_bounds__(128) void oproj_kernel(
    const bf16* __restrict__ Z, const float* __restrict__ Wo,
    const float* __restrict__ bo, const float* __restrict__ Xres,
    float* __restrict__ Y) {
  const int lane = threadIdx.x & 31;
  const int wave = threadIdx.x >> 5;
  const int n16  = lane & 15;
  const int half = lane >> 4;
  const int s0   = blockIdx.y * 64 + wave * 16;
  const int o0   = blockIdx.x * 64;

  v8f acc[4];
#pragma unroll
  for (int nt = 0; nt < 4; ++nt) acc[nt] = zero8();

  for (int k0 = 0; k0 < EDIM; k0 += 32) {
    v16bf a;
    const bf16* zp = Z + (size_t)(s0 + n16) * EDIM + k0;
#pragma unroll
    for (int i = 0; i < 8; ++i) {
      int k = a_kbase(i, half);
      a[2 * i]     = zp[k];
      a[2 * i + 1] = zp[k + 1];
    }
#pragma unroll
    for (int nt = 0; nt < 4; ++nt) {
      v16bf b;
      const float* wp = Wo + (size_t)(o0 + nt * 16 + n16) * EDIM + k0 + half * 16;
#pragma unroll
      for (int i = 0; i < 8; ++i) {
        b[2 * i]     = (bf16)wp[2 * i];
        b[2 * i + 1] = (bf16)wp[2 * i + 1];
      }
      acc[nt] = wmma_bf16(a, b, acc[nt]);
    }
  }

#pragma unroll
  for (int nt = 0; nt < 4; ++nt) {
    int o = o0 + nt * 16 + n16;
    float bv = bo[o];
#pragma unroll
    for (int r = 0; r < 8; ++r) {
      int s = s0 + r + 8 * half;
      Y[(size_t)s * EDIM + o] = acc[nt][r] + bv + Xres[(size_t)s * EDIM + o];
    }
  }
}

// ---------------------------------------------------------------------------
// Kernel 4: row LayerNorm.  256 threads per row (4 elems each).
// ---------------------------------------------------------------------------
__global__ __launch_bounds__(256) void ln_kernel(
    const float* __restrict__ Y, const float* __restrict__ g,
    const float* __restrict__ b, float* __restrict__ Out) {
  __shared__ float red0[8], red1[8];
  const int row = blockIdx.x;
  const float* y = Y + (size_t)row * EDIM;
  float v[4], s = 0.f, s2 = 0.f;
#pragma unroll
  for (int i = 0; i < 4; ++i) {
    v[i] = y[threadIdx.x + i * 256];
    s += v[i];
    s2 += v[i] * v[i];
  }
#pragma unroll
  for (int off = 1; off < 32; off <<= 1) {
    s  += __shfl_xor(s, off, 32);
    s2 += __shfl_xor(s2, off, 32);
  }
  const int wv = threadIdx.x >> 5;
  if ((threadIdx.x & 31) == 0) { red0[wv] = s; red1[wv] = s2; }
  __syncthreads();
  s = 0.f; s2 = 0.f;
#pragma unroll
  for (int i = 0; i < 8; ++i) { s += red0[i]; s2 += red1[i]; }
  const float mu  = s * (1.f / EDIM);
  const float var = s2 * (1.f / EDIM) - mu * mu;
  const float inv = rsqrtf(var + 1e-5f);
#pragma unroll
  for (int i = 0; i < 4; ++i) {
    int c = threadIdx.x + i * 256;
    Out[(size_t)row * EDIM + c] = (v[i] - mu) * inv * g[c] + b[c];
  }
}

// ---------------------------------------------------------------------------
// Launch: proj x3 -> attention -> out-proj(+residual) -> LayerNorm
// Workspace map (bytes): [0,8M) qh | [8M,16M) kh | [16M,24M) vh
//                        [24M,32M) attn-out bf16 | [32M,48M) y f32
// ---------------------------------------------------------------------------
extern "C" void kernel_launch(void* const* d_in, const int* in_sizes, int n_in,
                              void* d_out, int out_size, void* d_ws,
                              size_t ws_size, hipStream_t stream) {
  const float* q    = (const float*)d_in[0];
  const float* k    = (const float*)d_in[1];
  const float* v    = (const float*)d_in[2];
  const int*   mask = (const int*)d_in[3];
  const float* gp   = (const float*)d_in[4];
  const float* Wq   = (const float*)d_in[5];
  const float* bq   = (const float*)d_in[6];
  const float* Wk   = (const float*)d_in[7];
  const float* bk   = (const float*)d_in[8];
  const float* Wv   = (const float*)d_in[9];
  const float* bv   = (const float*)d_in[10];
  const float* Wo   = (const float*)d_in[11];
  const float* bo   = (const float*)d_in[12];
  const float* lng  = (const float*)d_in[13];
  const float* lnb  = (const float*)d_in[14];

  char* ws = (char*)d_ws;
  const size_t MB = 1024 * 1024;
  bf16*  qh = (bf16*)(ws);
  bf16*  kh = (bf16*)(ws + 8 * MB);
  bf16*  vh = (bf16*)(ws + 16 * MB);
  bf16*  zz = (bf16*)(ws + 24 * MB);
  float* yy = (float*)(ws + 32 * MB);

  dim3 gGemm(EDIM / 64, S_LEN / 64);   // (16, 64)
  proj_kernel<<<gGemm, 128, 0, stream>>>(q, Wq, bq, qh);
  proj_kernel<<<gGemm, 128, 0, stream>>>(k, Wk, bk, kh);
  proj_kernel<<<gGemm, 128, 0, stream>>>(v, Wv, bv, vh);

  dim3 gAttn(NHEAD, S_LEN / 64);       // heads fastest -> mask/gp L2 reuse
  attn_kernel<<<gAttn, 128, 0, stream>>>(qh, kh, vh, mask, gp, zz);

  oproj_kernel<<<gGemm, 128, 0, stream>>>(zz, Wo, bo, q, yy);
  ln_kernel<<<S_LEN, 256, 0, stream>>>(yy, lng, lnb, (float*)d_out);
}
